// LORAMultiheadAttention_46815143526810
// MI455X (gfx1250) — matrix operand order repaired and checked by reading, not compiled, not run on hardware
//
#include <hip/hip_runtime.h>
#include <cstdint>
#include <cstddef>

// ---------------------------------------------------------------------------
// Types for CDNA5 WMMA (wave32): bf16 16x16x32, f32 accumulate.
// ---------------------------------------------------------------------------
typedef float  v8f   __attribute__((ext_vector_type(8)));
typedef __bf16 v16bf __attribute__((ext_vector_type(16)));

union V16 { v16bf v; uint4 q[2]; };
union F8  { float4 f4[2]; float f[8]; };

static __device__ __forceinline__ v8f wmma_bf16(v16bf a, v16bf b, v8f c) {
  // (neg_a, A, neg_b, B, c_mod, C, reuse_a, reuse_b)
  return __builtin_amdgcn_wmma_f32_16x16x32_bf16(false, a, false, b, (short)0, c,
                                                 false, false);
}

static __device__ __forceinline__ __bf16 f2bf(float f) {
  union { float f; uint32_t u; } a; a.f = f;
  uint32_t u = a.u;
  uint32_t r = (u + 0x7FFFu + ((u >> 16) & 1u)) >> 16;  // RNE
  union { uint16_t s; __bf16 b; } o; o.s = (uint16_t)r; return o.b;
}
static __device__ __forceinline__ float bf2f(__bf16 b) {
  union { uint16_t s; __bf16 b; } i; i.b = b;
  union { uint32_t u; float f; } o; o.u = ((uint32_t)i.s) << 16; return o.f;
}
// fp32 -> bf16 hi + bf16 lo (hi+lo carries ~16 mantissa bits)
static __device__ __forceinline__ void split2(float x, __bf16& h, __bf16& l) {
  h = f2bf(x);
  l = f2bf(x - bf2f(h));
}

static __device__ __forceinline__ v8f vzero8() {
  v8f z;
#pragma unroll
  for (int i = 0; i < 8; ++i) z[i] = 0.0f;
  return z;
}

// Problem constants
#define E_DIM 1024
#define TOK   4096   // B*S = 4*1024
#define NH    16
#define HD    64
// GEMM block tile
#define BM 128
#define BN 128
#define BK 32
#define PLANE (BM * BK)  // bf16 elements per LDS plane (8KB)

// ---------------------------------------------------------------------------
// Kernel: fp32 -> (hi, lo) bf16 elementwise split
// ---------------------------------------------------------------------------
__global__ void split_f32_kernel(const float* __restrict__ src,
                                 __bf16* __restrict__ hi, __bf16* __restrict__ lo,
                                 int n) {
  int i = blockIdx.x * blockDim.x + threadIdx.x;
  if (i >= n) return;
  __bf16 h, l; split2(src[i], h, l);
  hi[i] = h; lo[i] = l;
}

// ---------------------------------------------------------------------------
// Kernel: W_eff = W + Blora @ Alora, output split into bf16 hi/lo.
// W: [E,E] row-major; Blora: [E,16]; Alora: [16,E].
// ---------------------------------------------------------------------------
__global__ void fold_lora_kernel(const float* __restrict__ W,
                                 const float* __restrict__ Blora,
                                 const float* __restrict__ Alora,
                                 __bf16* __restrict__ Whi, __bf16* __restrict__ Wlo) {
  int idx = blockIdx.x * blockDim.x + threadIdx.x;
  if (idx >= E_DIM * E_DIM) return;
  int o = idx >> 10, e = idx & 1023;
  float acc = W[idx];
#pragma unroll
  for (int r = 0; r < 16; ++r)
    acc = fmaf(Blora[o * 16 + r], Alora[r * E_DIM + e], acc);
  __bf16 h, l; split2(acc, h, l);
  Whi[idx] = h; Wlo[idx] = l;
}

// ---------------------------------------------------------------------------
// GEMM: out[T=4096, O=1024] = A[T,1024] * W[O,1024]^T + bias, bf16x3 split.
// LDS-staged, double-buffered 128x128 block tile; 256 threads = 8 waves
// (4 row-waves x 2 col-waves), each wave a 32x64 region (2x4 WMMA tiles).
// Per K=32 step: 24 WMMAs vs 24 ds_load_b128 + 8 staged global_load_b128.
// MODE 0: store fp32 to outF[t*1024+o]  (final o-projection)
// MODE 1: store bf16 hi/lo to Ohi/Olo[t*1024+o]  (q, k)
// MODE 2: store bf16 hi/lo transposed [b][h][d][s]  (v)
// ---------------------------------------------------------------------------
template <int MODE>
__global__ __launch_bounds__(256) void gemm_bf16x3(
    const __bf16* __restrict__ Ahi, const __bf16* __restrict__ Alo,
    const __bf16* __restrict__ Whi, const __bf16* __restrict__ Wlo,
    const float* __restrict__ bias,
    float* __restrict__ outF, __bf16* __restrict__ Ohi, __bf16* __restrict__ Olo) {
  // planes: 0=Ahi 1=Alo 2=Bhi 3=Blo, double buffered -> 64KB
  __shared__ alignas(16) __bf16 smem[2 * 4 * PLANE];

  const int tid  = threadIdx.x;
  const int wave = tid >> 5, lane = tid & 31;
  const int lr = lane & 15, g = lane >> 4;
  const int wrow = wave >> 1, wcol = wave & 1;
  const int rowBase = blockIdx.x * BM;
  const int colBase = blockIdx.y * BN;

  v8f c[2][4];
#pragma unroll
  for (int i = 0; i < 2; ++i)
#pragma unroll
    for (int j = 0; j < 4; ++j) c[i][j] = vzero8();

  // cooperative stage: 512 chunks of 16B per plane-pair layout
  // chunk -> (r = chunk>>2 row, cc = chunk&3 16B-slot within 64B row)
  auto stage_load = [&](int k0, uint4* regs) {
#pragma unroll
    for (int p = 0; p < 4; ++p) {
      const __bf16* src = (p == 0) ? Ahi : (p == 1) ? Alo : (p == 2) ? Whi : Wlo;
      const int base = (p < 2) ? rowBase : colBase;
#pragma unroll
      for (int j = 0; j < 2; ++j) {
        const int chunk = tid + j * 256;
        const int r = chunk >> 2, cc = chunk & 3;
        regs[p * 2 + j] =
            *(const uint4*)(src + (size_t)(base + r) * E_DIM + k0 + cc * 8);
      }
    }
  };
  auto stage_store = [&](int buf, const uint4* regs) {
#pragma unroll
    for (int p = 0; p < 4; ++p)
#pragma unroll
      for (int j = 0; j < 2; ++j) {
        const int chunk = tid + j * 256;
        const int r = chunk >> 2, cc = chunk & 3;
        *(uint4*)(smem + (size_t)(buf * 4 + p) * PLANE + r * BK + cc * 8) =
            regs[p * 2 + j];
      }
  };

  {
    uint4 regs[8];
    stage_load(0, regs);
    stage_store(0, regs);
  }
  __syncthreads();

  int cur = 0;
  for (int k0 = 0; k0 < E_DIM; k0 += BK) {
    uint4 regs[8];
    const bool more = (k0 + BK) < E_DIM;
    if (more) stage_load(k0 + BK, regs);  // overlap with compute below

    const __bf16* bufp = smem + (size_t)cur * 4 * PLANE;
    const __bf16* pAh = bufp;
    const __bf16* pAl = bufp + PLANE;
    const __bf16* pBh = bufp + 2 * PLANE;
    const __bf16* pBl = bufp + 3 * PLANE;

    V16 ah[2], al[2];
#pragma unroll
    for (int mt = 0; mt < 2; ++mt) {
      const int rloc = wrow * 32 + mt * 16 + lr;
      const __bf16* pa = pAh + rloc * BK;
      const __bf16* pl = pAl + rloc * BK;
      ah[mt].q[0] = *(const uint4*)(pa + g * 8);
      ah[mt].q[1] = *(const uint4*)(pa + 16 + g * 8);
      al[mt].q[0] = *(const uint4*)(pl + g * 8);
      al[mt].q[1] = *(const uint4*)(pl + 16 + g * 8);
    }
#pragma unroll
    for (int nt = 0; nt < 4; ++nt) {
      const int cloc = wcol * 64 + nt * 16 + lr;
      const __bf16* pb = pBh + cloc * BK + g * 8;
      const __bf16* pbl = pBl + cloc * BK + g * 8;
      V16 bh, bl;
      bh.q[0] = *(const uint4*)pb;   bh.q[1] = *(const uint4*)(pb + 16);
      bl.q[0] = *(const uint4*)pbl;  bl.q[1] = *(const uint4*)(pbl + 16);
#pragma unroll
      for (int mt = 0; mt < 2; ++mt) {
        c[mt][nt] = wmma_bf16(ah[mt].v, bh.v, c[mt][nt]);  // hi*hi
        c[mt][nt] = wmma_bf16(ah[mt].v, bl.v, c[mt][nt]);  // hi*lo
        c[mt][nt] = wmma_bf16(al[mt].v, bh.v, c[mt][nt]);  // lo*hi
      }
    }

    if (more) stage_store(cur ^ 1, regs);
    __syncthreads();
    cur ^= 1;
  }

#pragma unroll
  for (int mt = 0; mt < 2; ++mt)
#pragma unroll
    for (int nt = 0; nt < 4; ++nt) {
      const int n = colBase + wcol * 64 + nt * 16 + lr;
      const float bv = bias[n];
#pragma unroll
      for (int r = 0; r < 8; ++r) {
        const int m = rowBase + wrow * 32 + mt * 16 + r + 8 * g;
        const float v = c[mt][nt][r] + bv;
        if (MODE == 0) {
          outF[(size_t)m * E_DIM + n] = v;
        } else if (MODE == 1) {
          __bf16 h, l; split2(v, h, l);
          Ohi[(size_t)m * E_DIM + n] = h;
          Olo[(size_t)m * E_DIM + n] = l;
        } else {
          const int b = m >> 10, s = m & 1023;
          const int hh = n >> 6, d = n & 63;
          const size_t idx = (((size_t)(b * NH + hh) * HD + d) << 10) + s;
          __bf16 h, l; split2(v, h, l);
          Ohi[idx] = h; Olo[idx] = l;
        }
      }
    }
}

// ---------------------------------------------------------------------------
// Attention: one workgroup per (b, h, 16-row s-block). 128 threads = 4 waves.
// Phase 1: scores (16 x 1024) into LDS via bf16x3 WMMA, scaled by 1/8.
// Phase 2: per-row softmax (wave-parallel, shfl reductions).
// Phase 3: ctx = attn @ V (V stored transposed [b][h][d][s]) via bf16x3 WMMA.
// ---------------------------------------------------------------------------
__global__ __launch_bounds__(128) void attn_kernel(
    const __bf16* __restrict__ Qhi, const __bf16* __restrict__ Qlo,
    const __bf16* __restrict__ Khi, const __bf16* __restrict__ Klo,
    const __bf16* __restrict__ Vthi, const __bf16* __restrict__ Vtlo,
    __bf16* __restrict__ Chi, __bf16* __restrict__ Clo) {
  __shared__ alignas(16) float sc[16 * 1024];  // 64 KB score rows

  const int tid  = threadIdx.x;
  const int wave = tid >> 5, lane = tid & 31;
  const int lr = lane & 15, g = lane >> 4;
  const int s0 = blockIdx.x * 16;
  const int h  = blockIdx.y;
  const int b  = blockIdx.z;

  // --- load Q A-tiles (16 rows x 64 d) once: 2 k-steps, hi/lo ---
  V16 qh[2], ql[2];
#pragma unroll
  for (int kk = 0; kk < 2; ++kk) {
    const size_t base = (size_t)(b * 1024 + s0 + lr) * E_DIM + h * HD + kk * 32;
    const __bf16* pq = Qhi + base;
    const __bf16* pl = Qlo + base;
    qh[kk].q[0] = *(const uint4*)(pq + g * 8);
    qh[kk].q[1] = *(const uint4*)(pq + 16 + g * 8);
    ql[kk].q[0] = *(const uint4*)(pl + g * 8);
    ql[kk].q[1] = *(const uint4*)(pl + 16 + g * 8);
  }

  // --- Phase 1: each wave computes 16 t-tiles of scores ---
  for (int tt = wave * 16; tt < wave * 16 + 16; ++tt) {
    V16 kh[2], kl[2];
#pragma unroll
    for (int kk = 0; kk < 2; ++kk) {
      const size_t base =
          (size_t)(b * 1024 + tt * 16 + lr) * E_DIM + h * HD + kk * 32 + g * 8;
      const __bf16* pk = Khi + base;
      const __bf16* pl = Klo + base;
      kh[kk].q[0] = *(const uint4*)pk;  kh[kk].q[1] = *(const uint4*)(pk + 16);
      kl[kk].q[0] = *(const uint4*)pl;  kl[kk].q[1] = *(const uint4*)(pl + 16);
    }
    v8f c = vzero8();
#pragma unroll
    for (int kk = 0; kk < 2; ++kk) {
      c = wmma_bf16(qh[kk].v, kh[kk].v, c);
      c = wmma_bf16(qh[kk].v, kl[kk].v, c);
      c = wmma_bf16(ql[kk].v, kh[kk].v, c);
    }
#pragma unroll
    for (int r = 0; r < 8; ++r)
      sc[(r + 8 * g) * 1024 + tt * 16 + lr] = c[r] * 0.125f;  // 1/sqrt(64)
  }
  __syncthreads();

  // --- Phase 2: softmax, each wave owns 4 rows, 32 lanes stride the row ---
#pragma unroll
  for (int rr = 0; rr < 4; ++rr) {
    const int row = wave * 4 + rr;
    float* prow = &sc[row * 1024];
    float mx = -__builtin_inff();
    for (int j = 0; j < 32; ++j) mx = fmaxf(mx, prow[lane + j * 32]);
#pragma unroll
    for (int off = 16; off >= 1; off >>= 1) mx = fmaxf(mx, __shfl_xor(mx, off, 32));
    float sum = 0.0f;
    for (int j = 0; j < 32; ++j) {
      float e = __expf(prow[lane + j * 32] - mx);
      prow[lane + j * 32] = e;
      sum += e;
    }
#pragma unroll
    for (int off = 16; off >= 1; off >>= 1) sum += __shfl_xor(sum, off, 32);
    const float inv = 1.0f / sum;
    for (int j = 0; j < 32; ++j) prow[lane + j * 32] *= inv;
  }
  __syncthreads();

  // --- Phase 3: ctx[16, 64] = attn[16,1024] @ V[1024,64]; wave owns 16 d cols ---
  const int d = wave * 16 + lr;
  v8f c = vzero8();
  for (int ks = 0; ks < 32; ++ks) {
    const int k0 = ks * 32;
    // A from LDS (fp32 attn), convert to hi/lo on the fly
    V16 ah, al;
    {
      const float* pr = &sc[lr * 1024 + k0];
      F8 r0, r1;
      r0.f4[0] = *(const float4*)(pr + g * 8);
      r0.f4[1] = *(const float4*)(pr + g * 8 + 4);
      r1.f4[0] = *(const float4*)(pr + 16 + g * 8);
      r1.f4[1] = *(const float4*)(pr + 16 + g * 8 + 4);
#pragma unroll
      for (int j = 0; j < 8; ++j) {
        __bf16 h0, l0, h1, l1;
        split2(r0.f[j], h0, l0);
        split2(r1.f[j], h1, l1);
        ah.v[j] = h0;     al.v[j] = l0;
        ah.v[8 + j] = h1; al.v[8 + j] = l1;
      }
    }
    // B from V^T [b][h][d][s], t contiguous
    const size_t base = (((size_t)(b * NH + h) * HD + d) << 10) + k0 + g * 8;
    const __bf16* pv = Vthi + base;
    const __bf16* pl = Vtlo + base;
    V16 bh, bl;
    bh.q[0] = *(const uint4*)pv;  bh.q[1] = *(const uint4*)(pv + 16);
    bl.q[0] = *(const uint4*)pl;  bl.q[1] = *(const uint4*)(pl + 16);

    c = wmma_bf16(ah.v, bh.v, c);
    c = wmma_bf16(ah.v, bl.v, c);
    c = wmma_bf16(al.v, bh.v, c);
  }
#pragma unroll
  for (int r = 0; r < 8; ++r) {
    const int s = s0 + r + 8 * g;
    const size_t idx = (size_t)(b * 1024 + s) * E_DIM + h * HD + d;
    __bf16 hh, ll; split2(c[r], hh, ll);
    Chi[idx] = hh; Clo[idx] = ll;
  }
}

// ---------------------------------------------------------------------------
// Host launcher
// ---------------------------------------------------------------------------
extern "C" void kernel_launch(void* const* d_in, const int* in_sizes, int n_in,
                              void* d_out, int out_size, void* d_ws, size_t ws_size,
                              hipStream_t stream) {
  (void)in_sizes; (void)n_in; (void)out_size; (void)ws_size;
  const float* x    = (const float*)d_in[0];
  const float* qw   = (const float*)d_in[1];
  const float* qb   = (const float*)d_in[2];
  const float* qla  = (const float*)d_in[3];
  const float* qlb  = (const float*)d_in[4];
  const float* kw   = (const float*)d_in[5];
  const float* kb   = (const float*)d_in[6];
  const float* kla  = (const float*)d_in[7];
  const float* klb  = (const float*)d_in[8];
  const float* vw   = (const float*)d_in[9];
  const float* vb   = (const float*)d_in[10];
  const float* vla  = (const float*)d_in[11];
  const float* vlb  = (const float*)d_in[12];
  const float* ow   = (const float*)d_in[13];
  const float* ob   = (const float*)d_in[14];
  const float* ola  = (const float*)d_in[15];
  const float* olb  = (const float*)d_in[16];
  float* out = (float*)d_out;

  // Workspace carve-up (~96 MB)
  char* ws = (char*)d_ws;
  size_t off = 0;
  auto carve = [&](size_t bytes) { void* p = ws + off; off += bytes; return p; };
  const size_t XB = (size_t)TOK * E_DIM * sizeof(__bf16);   // 8 MB
  const size_t WB = (size_t)E_DIM * E_DIM * sizeof(__bf16); // 2 MB

  __bf16* xhi = (__bf16*)carve(XB);   __bf16* xlo = (__bf16*)carve(XB);
  __bf16* wqh = (__bf16*)carve(WB);   __bf16* wql = (__bf16*)carve(WB);
  __bf16* wkh = (__bf16*)carve(WB);   __bf16* wkl = (__bf16*)carve(WB);
  __bf16* wvh = (__bf16*)carve(WB);   __bf16* wvl = (__bf16*)carve(WB);
  __bf16* woh = (__bf16*)carve(WB);   __bf16* wol = (__bf16*)carve(WB);
  __bf16* qhi = (__bf16*)carve(XB);   __bf16* qlo = (__bf16*)carve(XB);
  __bf16* khi = (__bf16*)carve(XB);   __bf16* klo = (__bf16*)carve(XB);
  __bf16* vth = (__bf16*)carve(XB);   __bf16* vtl = (__bf16*)carve(XB);
  __bf16* chi = (__bf16*)carve(XB);   __bf16* clo = (__bf16*)carve(XB);

  // 1) split x into hi/lo bf16
  split_f32_kernel<<<(TOK * E_DIM) / 256, 256, 0, stream>>>(x, xhi, xlo, TOK * E_DIM);

  // 2) fold LoRA into effective weights (W + B@A), split to bf16 hi/lo
  const int foldGrid = (E_DIM * E_DIM) / 256;
  fold_lora_kernel<<<foldGrid, 256, 0, stream>>>(qw, qlb, qla, wqh, wql);
  fold_lora_kernel<<<foldGrid, 256, 0, stream>>>(kw, klb, kla, wkh, wkl);
  fold_lora_kernel<<<foldGrid, 256, 0, stream>>>(vw, vlb, vla, wvh, wvl);
  fold_lora_kernel<<<foldGrid, 256, 0, stream>>>(ow, olb, ola, woh, wol);

  // 3) projections (128x128 LDS-staged block GEMM)
  dim3 ggrid(TOK / BM, E_DIM / BN);
  gemm_bf16x3<1><<<ggrid, 256, 0, stream>>>(xhi, xlo, wqh, wql, qb,
                                            nullptr, qhi, qlo);
  gemm_bf16x3<1><<<ggrid, 256, 0, stream>>>(xhi, xlo, wkh, wkl, kb,
                                            nullptr, khi, klo);
  gemm_bf16x3<2><<<ggrid, 256, 0, stream>>>(xhi, xlo, wvh, wvl, vb,
                                            nullptr, vth, vtl);

  // 4) attention: grid (s-blocks, heads, batch)
  attn_kernel<<<dim3(1024 / 16, NH, 4), 128, 0, stream>>>(qhi, qlo, khi, klo,
                                                          vth, vtl, chi, clo);

  // 5) output projection -> fp32 result
  gemm_bf16x3<0><<<ggrid, 256, 0, stream>>>(chi, clo, woh, wol, ob,
                                            out, nullptr, nullptr);
}
